// SpatialLayer_15530601742655
// MI455X (gfx1250) — matrix operand order repaired and checked by reading, not compiled, not run on hardware
//
#include <hip/hip_runtime.h>

// ---------------------------------------------------------------------------
// Graph attention layer for MI455X (gfx1250, wave32, WMMA bf16 16x16x32).
//
//  * KW[r] = rel_w[r] @ k_w / sqrt(U),  VW[r] = rel_w[r] @ v_w  precomputed,
//    so the per-edge relation transform is folded into K/V (2 GEMMs, not 3).
//  * Per-row weight selection (entity / relation id) is done by masking the
//    WMMA A-fragment per lane (lane == row mod 16) and looping the 8 ids.
//    BOTH masked fragments of a pair are computed BEFORE either WMMA issues:
//    overlapping live ranges force disjoint VGPR sets, so the WMMA->VALU WAR
//    hazard (4 co-exec slots for bf16 WMMA) is covered by the other WMMA and
//    the next mask block instead of compiler-inserted v_nops.
//  * All weights pre-swizzled to B-fragment layout (32B contiguous per lane).
//  * center table stored bf16 with a zero pad row 0 -> neighbor gather is a
//    straight row copy, fully L2-resident (4.2 MB table vs 192 MB L2).
// ---------------------------------------------------------------------------

#define N_NODES 16384
#define DEG     32
#define INPU    256
#define UU      128
#define RELS    8
#define ENTS    8

typedef __attribute__((ext_vector_type(16))) __bf16 v16bf;
typedef __attribute__((ext_vector_type(8)))  float  v8f;

union BfFrag { v16bf v; unsigned short h[16]; uint4 q4[2]; };

__device__ __forceinline__ unsigned short f2bf(float f){
  unsigned int u = __builtin_bit_cast(unsigned int, f);
  unsigned int r = u + 0x7FFFu + ((u >> 16) & 1u);   // round-to-nearest-even
  return (unsigned short)(r >> 16);
}

// element (k,n) of matrix `mat` -> flat index in B-fragment-swizzled storage.
// B frag (32x16 bf16): lanes 0-15 hold col n=lane, K=ks*32+0..15 (j=0..15);
// lanes 16-31 hold col n=lane-16, K=ks*32+16..31.
__device__ __forceinline__ size_t fidx_elem(int mat, int k, int n, int nks){
  int ntile = n >> 4, nl = n & 15;
  int ks = k >> 5,  kk = k & 31;
  int lane = nl + ((kk & 16) ? 16 : 0);
  int j = kk & 15;
  return ((((size_t)mat * 8 + ntile) * nks + ks) * 32 + lane) * 16 + j;
}
__device__ __forceinline__ size_t fidx_frag(int mat, int ntile, int ks, int nks, int lane){
  return ((((size_t)mat * 8 + ntile) * nks + ks) * 32 + lane) * 16;
}

// ------------------------- weight preparation ------------------------------

__global__ void k_build_pw(const float* __restrict__ pw, unsigned short* __restrict__ pwf){
  int gid = blockIdx.x * blockDim.x + threadIdx.x;   // ENTS*INPU*UU = 262144
  int e = gid >> 15;
  int rem = gid & 32767;
  int k = rem >> 7, n = rem & 127;
  pwf[fidx_elem(e, k, n, 8)] = f2bf(pw[gid]);
}

__global__ void k_build_qw(const float* __restrict__ qw, unsigned short* __restrict__ qwf,
                           float scale){
  int gid = blockIdx.x * blockDim.x + threadIdx.x;   // UU*UU = 16384
  int k = gid >> 7, n = gid & 127;
  qwf[fidx_elem(0, k, n, 4)] = f2bf(qw[gid] * scale);
}

__global__ void k_build_kvw(const float* __restrict__ rw, const float* __restrict__ kw,
                            const float* __restrict__ vw, unsigned short* __restrict__ kwf,
                            unsigned short* __restrict__ vwf, float scale){
  int gid = blockIdx.x * blockDim.x + threadIdx.x;   // 2*RELS*UU*UU = 262144
  int m = gid >> 17;
  int rem = gid & 131071;
  int r = rem >> 14;
  int rem2 = rem & 16383;
  int k = rem2 >> 7, n = rem2 & 127;
  const float* rrow = rw + ((size_t)r * 128 + k) * 128;
  const float* wcol = (m == 0 ? kw : vw) + n;
  float s = 0.f;
  for (int u2 = 0; u2 < 128; ++u2) s += rrow[u2] * wcol[(size_t)u2 * 128];
  if (m == 0) kwf[fidx_elem(r, k, n, 4)] = f2bf(s * scale);
  else        vwf[fidx_elem(r, k, n, 4)] = f2bf(s);
}

// ------------------------- center + q kernel -------------------------------
// 16 nodes / block, 8 waves, wave w owns column tile w (16 cols of 128).

__global__ void k_center(const float* __restrict__ ns, const int* __restrict__ pe_g,
                         const unsigned short* __restrict__ pwf,
                         const unsigned short* __restrict__ qwf,
                         float* __restrict__ c32, unsigned short* __restrict__ cbf_g,
                         float* __restrict__ q32){
  __shared__ __align__(16) unsigned short xs[16 * 256];   // node_state tile bf16
  __shared__ __align__(16) unsigned short cb[16 * 128];   // center tile bf16
  __shared__ int pe[16];
  int tid = threadIdx.x;
  int n0 = blockIdx.x * 16;

  if (blockIdx.x == 0 && tid < 64)            // zero the pad row (row 0)
    reinterpret_cast<unsigned int*>(cbf_g)[tid] = 0u;

  for (int i = tid; i < 16 * 256; i += 256) xs[i] = f2bf(ns[(size_t)n0 * 256 + i]);
  if (tid < 16) pe[tid] = pe_g[n0 + tid];
  __syncthreads();

  int w = tid >> 5, l = tid & 31;
  int ml = l & 15;
  int kb = (l < 16) ? 0 : 16;    // A frag K sub-offset (contiguous K, matches B frag)
  int hi = (l < 16) ? 0 : 8;     // C frag row offset
  uint4 z4 = make_uint4(0, 0, 0, 0);
  int myid = pe[ml];

  v8f c = {};
  for (int ks = 0; ks < 8; ++ks){
    BfFrag a;
    const unsigned short* xr = xs + ml * 256 + ks * 32 + kb;
    a.q4[0] = *reinterpret_cast<const uint4*>(xr);
    a.q4[1] = *reinterpret_cast<const uint4*>(xr + 8);
    #pragma unroll
    for (int e = 0; e < 8; e += 2){
      BfFrag b0, b1;
      const uint4* bp0 = reinterpret_cast<const uint4*>(pwf + fidx_frag(e,     w, ks, 8, l));
      const uint4* bp1 = reinterpret_cast<const uint4*>(pwf + fidx_frag(e + 1, w, ks, 8, l));
      b0.q4[0] = bp0[0]; b0.q4[1] = bp0[1];
      b1.q4[0] = bp1[0]; b1.q4[1] = bp1[1];
      // compute BOTH masked fragments first -> overlapping live ranges ->
      // disjoint registers; WMMAs then cover each other's WAR hazard window
      bool on0 = (myid == e);
      bool on1 = (myid == e + 1);
      BfFrag am0, am1;
      am0.q4[0] = on0 ? a.q4[0] : z4; am0.q4[1] = on0 ? a.q4[1] : z4;
      am1.q4[0] = on1 ? a.q4[0] : z4; am1.q4[1] = on1 ? a.q4[1] : z4;
      c = __builtin_amdgcn_wmma_f32_16x16x32_bf16(false, am0.v, false, b0.v,
                                                  (short)0, c, false, false);
      c = __builtin_amdgcn_wmma_f32_16x16x32_bf16(false, am1.v, false, b1.v,
                                                  (short)0, c, false, false);
    }
  }
  int col = w * 16 + ml;
  #pragma unroll
  for (int i = 0; i < 8; ++i){
    int m = hi + i;
    float fv = c[i];
    unsigned short hb = f2bf(fv);
    c32[(size_t)(n0 + m) * 128 + col] = fv;
    cbf_g[(size_t)(n0 + m + 1) * 128 + col] = hb;   // +1: padded table
    cb[m * 128 + col] = hb;
  }
  __syncthreads();

  // q = center @ (q_w / sqrt(U))
  v8f qc = {};
  for (int ks = 0; ks < 4; ++ks){
    BfFrag a;
    const unsigned short* xr = cb + ml * 128 + ks * 32 + kb;
    a.q4[0] = *reinterpret_cast<const uint4*>(xr);
    a.q4[1] = *reinterpret_cast<const uint4*>(xr + 8);
    BfFrag b;
    const uint4* bp = reinterpret_cast<const uint4*>(qwf + fidx_frag(0, w, ks, 4, l));
    b.q4[0] = bp[0]; b.q4[1] = bp[1];
    qc = __builtin_amdgcn_wmma_f32_16x16x32_bf16(false, a.v, false, b.v,
                                                 (short)0, qc, false, false);
  }
  #pragma unroll
  for (int i = 0; i < 8; ++i)
    q32[(size_t)(n0 + hi + i) * 128 + col] = qc[i];
}

// ------------------------- attention kernel --------------------------------
// 4 nodes / block (128 edges). Dynamic LDS:
//   xls  : bf16 [4][32][128]  32768 B   gathered neighbor embeddings
//   kls  : f32  [4][32][128]  65536 B   K matrix
//   qls  : f32  [4][128]       2048 B
//   aggs : f32  [4][128]       2048 B
//   sc   : f32  [4][32]         512 B   scores -> attn
//   rels : i32  [4][32]         512 B
#define ATTN_SMEM (32768 + 65536 + 2048 + 2048 + 512 + 512)

__global__ void k_attn(const int* __restrict__ adj, const int* __restrict__ rel_g,
                       const unsigned short* __restrict__ cbf_g,
                       const float* __restrict__ c32, const float* __restrict__ q32,
                       const unsigned short* __restrict__ kwf,
                       const unsigned short* __restrict__ vwf,
                       const float* __restrict__ fcw, const float* __restrict__ fcb,
                       float* __restrict__ out){
  extern __shared__ __align__(16) char smem[];
  unsigned short* xls = (unsigned short*)smem;
  float* kls  = (float*)(smem + 32768);
  float* qls  = (float*)(smem + 98304);
  float* aggs = (float*)(smem + 100352);
  float* sc   = (float*)(smem + 102400);
  int*   rels = (int*)  (smem + 102912);

  int tid = threadIdx.x;
  int n0 = blockIdx.x * 4;

  // gather: 128 edge rows x 128 bf16; 2 threads per row, 128B each
  {
    int row = tid >> 1, half = tid & 1;
    int a = adj[n0 * 32 + row];                 // 0 == pad -> zero row
    const uint4* src = reinterpret_cast<const uint4*>(cbf_g + (size_t)a * 128 + half * 64);
    uint4* dst = reinterpret_cast<uint4*>(xls + row * 128 + half * 64);
    #pragma unroll
    for (int i = 0; i < 8; ++i) dst[i] = src[i];
    if (tid < 128) rels[tid] = rel_g[n0 * 32 + tid];
  }
  for (int i = tid; i < 512; i += 256) qls[i] = q32[(size_t)n0 * 128 + i];
  __syncthreads();

  int w = tid >> 5, l = tid & 31;
  int ml = l & 15;
  int kb = (l < 16) ? 0 : 16;
  int hi = (l < 16) ? 0 : 8;
  uint4 z4 = make_uint4(0, 0, 0, 0);

  // ---- K = x @ KW[rel] (masked per relation) ----
  v8f kacc[4][2] = {};
  for (int ks = 0; ks < 4; ++ks){
    BfFrag a[4][2]; int rm[4][2];
    #pragma unroll
    for (int nn = 0; nn < 4; ++nn)
      #pragma unroll
      for (int h = 0; h < 2; ++h){
        int edge = h * 16 + ml;
        const unsigned short* xr = xls + (nn * 32 + edge) * 128 + ks * 32 + kb;
        a[nn][h].q4[0] = *reinterpret_cast<const uint4*>(xr);
        a[nn][h].q4[1] = *reinterpret_cast<const uint4*>(xr + 8);
        rm[nn][h] = rels[nn * 32 + edge];
      }
    for (int r = 0; r < 8; ++r){
      BfFrag b;
      const uint4* bp = reinterpret_cast<const uint4*>(kwf + fidx_frag(r, w, ks, 4, l));
      b.q4[0] = bp[0]; b.q4[1] = bp[1];
      #pragma unroll
      for (int nn = 0; nn < 4; ++nn){
        bool on0 = (rm[nn][0] == r);
        bool on1 = (rm[nn][1] == r);
        BfFrag am0, am1;          // both masks before either WMMA
        am0.q4[0] = on0 ? a[nn][0].q4[0] : z4;
        am0.q4[1] = on0 ? a[nn][0].q4[1] : z4;
        am1.q4[0] = on1 ? a[nn][1].q4[0] : z4;
        am1.q4[1] = on1 ? a[nn][1].q4[1] : z4;
        kacc[nn][0] = __builtin_amdgcn_wmma_f32_16x16x32_bf16(
            false, am0.v, false, b.v, (short)0, kacc[nn][0], false, false);
        kacc[nn][1] = __builtin_amdgcn_wmma_f32_16x16x32_bf16(
            false, am1.v, false, b.v, (short)0, kacc[nn][1], false, false);
      }
    }
  }
  #pragma unroll
  for (int nn = 0; nn < 4; ++nn)
    #pragma unroll
    for (int h = 0; h < 2; ++h)
      #pragma unroll
      for (int i = 0; i < 8; ++i){
        int d = h * 16 + hi + i;
        kls[(nn * 32 + d) * 128 + w * 16 + ml] = kacc[nn][h][i];
      }

  // ---- V = x @ VW[rel] (kept in registers) ----
  v8f vacc[4][2] = {};
  for (int ks = 0; ks < 4; ++ks){
    BfFrag a[4][2]; int rm[4][2];
    #pragma unroll
    for (int nn = 0; nn < 4; ++nn)
      #pragma unroll
      for (int h = 0; h < 2; ++h){
        int edge = h * 16 + ml;
        const unsigned short* xr = xls + (nn * 32 + edge) * 128 + ks * 32 + kb;
        a[nn][h].q4[0] = *reinterpret_cast<const uint4*>(xr);
        a[nn][h].q4[1] = *reinterpret_cast<const uint4*>(xr + 8);
        rm[nn][h] = rels[nn * 32 + edge];
      }
    for (int r = 0; r < 8; ++r){
      BfFrag b;
      const uint4* bp = reinterpret_cast<const uint4*>(vwf + fidx_frag(r, w, ks, 4, l));
      b.q4[0] = bp[0]; b.q4[1] = bp[1];
      #pragma unroll
      for (int nn = 0; nn < 4; ++nn){
        bool on0 = (rm[nn][0] == r);
        bool on1 = (rm[nn][1] == r);
        BfFrag am0, am1;
        am0.q4[0] = on0 ? a[nn][0].q4[0] : z4;
        am0.q4[1] = on0 ? a[nn][0].q4[1] : z4;
        am1.q4[0] = on1 ? a[nn][1].q4[0] : z4;
        am1.q4[1] = on1 ? a[nn][1].q4[1] : z4;
        vacc[nn][0] = __builtin_amdgcn_wmma_f32_16x16x32_bf16(
            false, am0.v, false, b.v, (short)0, vacc[nn][0], false, false);
        vacc[nn][1] = __builtin_amdgcn_wmma_f32_16x16x32_bf16(
            false, am1.v, false, b.v, (short)0, vacc[nn][1], false, false);
      }
    }
  }
  __syncthreads();

  // ---- scores + softmax (attn over 32 neighbors, pad edges masked) ----
  {
    // 2 threads per (node, neighbor): 64-element partial dot + lane-pair sum
    int pr = tid >> 1, half = tid & 1;       // pr = nn*32 + d
    int nn = pr >> 5, d = pr & 31;
    const float* qv = qls + nn * 128 + half * 64;
    const float* kv = kls + (nn * 32 + d) * 128 + half * 64;
    float s = 0.f;
    for (int c2 = 0; c2 < 64; ++c2) s += qv[c2] * kv[c2];
    s += __shfl_xor(s, 1, 32);
    if (half == 0) sc[pr] = (rels[pr] == 0) ? -1e9f : s;
  }
  __syncthreads();
  if (tid < 4){
    float mx = -3.0e38f;
    for (int d = 0; d < 32; ++d) mx = fmaxf(mx, sc[tid * 32 + d]);
    float sum = 0.f;
    for (int d = 0; d < 32; ++d){
      float e = __expf(sc[tid * 32 + d] - mx);
      sc[tid * 32 + d] = e; sum += e;
    }
    float inv = 1.f / sum;
    for (int d = 0; d < 32; ++d) sc[tid * 32 + d] *= inv;
  }
  __syncthreads();

  // ---- agg = attn . V ; reduce lane-halves of the C fragment via shfl ----
  #pragma unroll
  for (int nn = 0; nn < 4; ++nn){
    float p = 0.f;
    #pragma unroll
    for (int h = 0; h < 2; ++h){
      int dbase = h * 16 + hi;
      #pragma unroll
      for (int i = 0; i < 8; ++i) p += sc[nn * 32 + dbase + i] * vacc[nn][h][i];
    }
    p += __shfl_xor(p, 16, 32);
    if (l < 16) aggs[nn * 128 + w * 16 + ml] = p;
  }
  __syncthreads();

  // ---- out = center + relu(agg @ fc_w + fc_b) ----
  for (int o = tid; o < 512; o += 256){
    int nn = o >> 7, c2 = o & 127;
    float s = fcb[c2];
    const float* av = aggs + nn * 128;
    for (int k2 = 0; k2 < 128; ++k2) s += av[k2] * fcw[(size_t)k2 * 128 + c2];
    size_t oi = (size_t)(n0 + nn) * 128 + c2;
    out[oi] = c32[oi] + fmaxf(s, 0.f);
  }
}

// ------------------------- launcher ---------------------------------------

extern "C" void kernel_launch(void* const* d_in, const int* in_sizes, int n_in,
                              void* d_out, int out_size, void* d_ws, size_t ws_size,
                              hipStream_t stream){
  const float* node_state     = (const float*)d_in[0];
  const int*   adjacency      = (const int*)  d_in[1];
  const int*   point_enc      = (const int*)  d_in[2];
  const int*   relation_enc   = (const int*)  d_in[3];
  const float* point_enc_w    = (const float*)d_in[4];
  const float* relation_enc_w = (const float*)d_in[5];
  const float* q_w            = (const float*)d_in[6];
  const float* k_w            = (const float*)d_in[7];
  const float* v_w            = (const float*)d_in[8];
  const float* fc_w           = (const float*)d_in[9];
  const float* fc_b           = (const float*)d_in[10];
  float* out = (float*)d_out;

  char* ws = (char*)d_ws;
  size_t off = 0;
  auto alloc = [&](size_t bytes) -> char* {
    char* p = ws + off;
    off = (off + bytes + 255) & ~(size_t)255;
    return p;
  };
  unsigned short* pwf = (unsigned short*)alloc((size_t)ENTS * INPU * UU * 2);     // 512 KB
  unsigned short* qwf = (unsigned short*)alloc((size_t)UU * UU * 2);              //  32 KB
  unsigned short* kwf = (unsigned short*)alloc((size_t)RELS * UU * UU * 2);       // 256 KB
  unsigned short* vwf = (unsigned short*)alloc((size_t)RELS * UU * UU * 2);       // 256 KB
  unsigned short* cbf = (unsigned short*)alloc((size_t)(N_NODES + 1) * UU * 2);   // 4.2 MB
  float* c32 = (float*)alloc((size_t)N_NODES * UU * 4);                           // 8 MB
  float* q32 = (float*)alloc((size_t)N_NODES * UU * 4);                           // 8 MB
  (void)ws_size; (void)in_sizes; (void)n_in; (void)out_size;

  const float inv_scale = 0.08838834764831845f;  // 1/sqrt(128)

  k_build_pw <<<(ENTS * INPU * UU) / 256, 256, 0, stream>>>(point_enc_w, pwf);
  k_build_qw <<<(UU * UU) / 256,           256, 0, stream>>>(q_w, qwf, inv_scale);
  k_build_kvw<<<(2 * RELS * UU * UU) / 256, 256, 0, stream>>>(relation_enc_w, k_w, v_w,
                                                              kwf, vwf, inv_scale);
  k_center<<<N_NODES / 16, 256, 0, stream>>>(node_state, point_enc, pwf, qwf,
                                             c32, cbf, q32);
  k_attn<<<N_NODES / 4, 256, ATTN_SMEM, stream>>>(adjacency, relation_enc, cbf, c32,
                                                  q32, kwf, vwf, fc_w, fc_b, out);
}
